// ComplexProjMeasurement_13709535609528
// MI455X (gfx1250) — matrix operand order repaired and checked by reading, not compiled, not run on hardware
//
#include <hip/hip_runtime.h>

#define DIM  512
#define KTOT 1024
#define NB   (DIM * KTOT)   // 524288 elements per B operand

typedef __attribute__((ext_vector_type(16))) __bf16 v16bf;
typedef __attribute__((ext_vector_type(8)))  float  v8f;
typedef unsigned int v4ui __attribute__((ext_vector_type(4)));
typedef int          v8i  __attribute__((ext_vector_type(8)));
typedef int          v4i  __attribute__((ext_vector_type(4)));

static __device__ __forceinline__ v8f wmma_bf16(v16bf a, v16bf b, v8f c) {
  // 8 args: (neg_a, A, neg_b, B, c_mod, C, reuse_a, reuse_b)
  return __builtin_amdgcn_wmma_f32_16x16x32_bf16(false, a, false, b, (short)0, c, false, false);
}

// Issue one TDM load: D# group0 rebuilt per call (lds/global addr), groups 1-3 cached.
// This toolchain's builtin takes 6 args: (g0, g1, g2, g3, g4, cpol).
static __device__ __forceinline__ void tdm_load_tile(unsigned int lds_addr,
                                                     unsigned long long gaddr,
                                                     v8i g1, v4i g2, v4i g3) {
  v4ui g0;
  g0.x = 1u;                                               // count=1, user D#, no gather
  g0.y = lds_addr;                                         // LDS byte address
  g0.z = (unsigned int)(gaddr & 0xFFFFFFFFu);              // global_addr[31:0]
  g0.w = (unsigned int)((gaddr >> 32) & 0x1FFFFFFu)        // global_addr[56:32]
       | (2u << 30);                                       // type=2 ("image")
  v8i g4 = {0, 0, 0, 0, 0, 0, 0, 0};                       // unused extra group
  __builtin_amdgcn_tensor_load_to_lds(g0, g1, g2, g3, g4, 0);
}

// Build hi/lo bf16 split of the two K=1024 B operands (stored [i][k], k contiguous):
//   BP[i][k] = (k<512) ?  Kr[i][k]  :  Ki[i][k-512]      (weighted later by kr[i,a])
//   BQ[i][k] = (k<512) ?  Ki[i][k]  : -Kr[i][k-512]      (weighted later by ki[i,a])
__global__ void prep_kernel(const float* __restrict__ kern,
                            __bf16* __restrict__ bph, __bf16* __restrict__ bpl,
                            __bf16* __restrict__ bqh, __bf16* __restrict__ bql) {
  int idx = blockIdx.x * blockDim.x + threadIdx.x;
  if (idx >= NB) return;
  int i = idx >> 10;
  int k = idx & (KTOT - 1);
  int b = k & (DIM - 1);
  const float2* k2 = (const float2*)kern;       // kernel[(i*512+b)*2 + {0,1}]
  float2 w = k2[(size_t)i * DIM + b];           // {kr, ki}
  float pv = (k < DIM) ? w.x : w.y;
  float qv = (k < DIM) ? w.y : -w.x;
  __bf16 ph = (__bf16)pv;
  __bf16 qh = (__bf16)qv;
  bph[idx] = ph;  bpl[idx] = (__bf16)(pv - (float)ph);
  bqh[idx] = qh;  bql[idx] = (__bf16)(qv - (float)qh);
}

// One workgroup: batch j, i-tile of 64, full a-range (512), K=1024.
// 8 waves; wave w owns a in [64w, 64w+64). B panels arrive via TDM (double-buffered LDS).
__global__ __launch_bounds__(256) void quad_kernel(
    const float* __restrict__ in_real, const float* __restrict__ in_imag,
    const float* __restrict__ kern,
    const __bf16* __restrict__ bmats,   // [mat=4][i=512][k=1024] contiguous
    float* __restrict__ out) {
  __shared__ __bf16 ldsB[2][4 * 64 * 32];   // 2 x 16 KB: {BPh,BPl,BQh,BQl} x (64 n x 32 k)
  __shared__ float  red[8][64];

  const int j    = blockIdx.y;
  const int i0   = blockIdx.x * 64;
  const int t    = threadIdx.x;
  const int lane = t & 31;
  const int wave = t >> 5;
  const int a0   = wave * 64;
  const int lrow = lane & 15;
  const int lhi  = lane >> 4;

  v8f accP[4][4], accQ[4][4];
  #pragma unroll
  for (int mt = 0; mt < 4; ++mt)
    #pragma unroll
    for (int nt = 0; nt < 4; ++nt)
      #pragma unroll
      for (int e = 0; e < 8; ++e) { accP[mt][nt][e] = 0.f; accQ[mt][nt][e] = 0.f; }

  const float* __restrict__ baseR = in_real + (size_t)j * DIM * DIM;
  const float* __restrict__ baseI = in_imag + (size_t)j * DIM * DIM;

  // --- Tensor DMA descriptor, constant groups -------------------------------
  // 3D tensor: X=k (len 1024, elem 2B), Y=i (len 512, stride 1024 elems),
  //            Z=mat (len 4, stride 524288 elems). Tile = 32 x 64 x 4 = 16 KB,
  //            written linearly to LDS in [mat][n][k] order.
  v8i g1;
  g1[0] = (int)(1u << 16);                         // wg_mask=0 (no cluster), data_size=2B
  g1[1] = (int)((KTOT & 0xFFFF) << 16);            // tensor_dim0[15:0]  (bits 63:48)
  g1[2] = (int)((KTOT >> 16) | ((DIM & 0xFFFF) << 16)); // dim0[31:16] | dim1[15:0]
  g1[3] = (int)((DIM >> 16) | (32u << 16));        // dim1[31:16] | tile_dim0=32
  g1[4] = (int)(64u | (4u << 16));                 // tile_dim1=64 | tile_dim2=4
  g1[5] = (int)KTOT;                               // tensor_dim0_stride = 1024 (48b lo32)
  g1[6] = 0;                                       // stride0 hi16 | stride1[15:0] (524288 lo16=0)
  g1[7] = (int)(NB >> 16);                         // tensor_dim1_stride[47:16] = 8
  v4i g2 = {4, 0, 0, 0};                           // tensor_dim2 = 4 (mat)
  v4i g3 = {0, 0, 0, 0};

  const unsigned long long gbase =
      (unsigned long long)(uintptr_t)bmats + (unsigned long long)i0 * KTOT * 2u;

  if (wave == 0)   // prologue: fill buffer 0 for k=0
    tdm_load_tile((unsigned int)(uintptr_t)&ldsB[0][0], gbase, g1, g2, g3);

  #pragma unroll 1
  for (int k = 0; k < KTOT; k += 32) {
    const int cur = (k >> 5) & 1;
    if (wave == 0) {
      if (k + 32 < KTOT) {
        tdm_load_tile((unsigned int)(uintptr_t)&ldsB[cur ^ 1][0],
                      gbase + (unsigned long long)(k + 32) * 2u, g1, g2, g3);
        __builtin_amdgcn_s_wait_tensorcnt(1);   // current buffer's TDM done
      } else {
        __builtin_amdgcn_s_wait_tensorcnt(0);
      }
    }
    __syncthreads();   // buffer 'cur' visible to all waves

    // B fragments: lane holds column n = lrow, contiguous K-chunk lhi*16..+15
    v16bf bf[4][4];   // [nt][mat]
    #pragma unroll
    for (int nt = 0; nt < 4; ++nt)
      #pragma unroll
      for (int m = 0; m < 4; ++m) {
        const __bf16* p = &ldsB[cur][((m * 64) + nt * 16 + lrow) * 32 + lhi * 16];
        #pragma unroll
        for (int e = 0; e < 16; ++e) bf[nt][m][e] = p[e];
      }

    // A = [R_j | I_j] along K
    const float* __restrict__ Asrc = (k < DIM) ? (baseR + k) : (baseI + (k - DIM));
    #pragma unroll
    for (int mt = 0; mt < 4; ++mt) {
      // A 16x32 bf16 layout: lane m: K (lhi*16..+15) contiguous
      const float* __restrict__ p = Asrc + (size_t)(a0 + mt * 16 + lrow) * DIM + lhi * 16;
      float4 x0 = *(const float4*)(p + 0);
      float4 x1 = *(const float4*)(p + 4);
      float4 x2 = *(const float4*)(p + 8);
      float4 x3 = *(const float4*)(p + 12);
      float xs[16] = {x0.x, x0.y, x0.z, x0.w, x1.x, x1.y, x1.z, x1.w,
                      x2.x, x2.y, x2.z, x2.w, x3.x, x3.y, x3.z, x3.w};
      v16bf ah, al;
      #pragma unroll
      for (int e = 0; e < 16; ++e) {
        __bf16 h = (__bf16)xs[e];
        ah[e] = h;
        al[e] = (__bf16)(xs[e] - (float)h);
      }
      #pragma unroll
      for (int nt = 0; nt < 4; ++nt) {
        v8f cP = accP[mt][nt], cQ = accQ[mt][nt];
        cP = wmma_bf16(ah, bf[nt][0], cP);   // hi*hi
        cP = wmma_bf16(al, bf[nt][0], cP);   // lo*hi
        cP = wmma_bf16(ah, bf[nt][1], cP);   // hi*lo
        cQ = wmma_bf16(ah, bf[nt][2], cQ);
        cQ = wmma_bf16(al, bf[nt][2], cQ);
        cQ = wmma_bf16(ah, bf[nt][3], cQ);
        accP[mt][nt] = cP; accQ[mt][nt] = cQ;
      }
    }
    __syncthreads();   // all reads of buffer 'cur' done; TDM may overwrite next iter
  }

  // out[j,i] = sum_a kr[i,a]*P[a,i] + ki[i,a]*Q[a,i]
  // C/D layout: element (m,n): m = r + 8*lhi, n = lrow
  const float2* __restrict__ k2 = (const float2*)kern;
  #pragma unroll
  for (int nt = 0; nt < 4; ++nt) {
    const int ig = i0 + nt * 16 + lrow;
    float partial = 0.f;
    #pragma unroll
    for (int mt = 0; mt < 4; ++mt) {
      const int ab = a0 + mt * 16 + lhi * 8;
      #pragma unroll
      for (int r = 0; r < 8; ++r) {
        float2 w = k2[(size_t)ig * DIM + ab + r];   // {kr[i,a], ki[i,a]}
        partial += accP[mt][nt][r] * w.x + accQ[mt][nt][r] * w.y;
      }
    }
    partial += __shfl_xor(partial, 16, 32);   // merge the two half-wave M groups
    if (lhi == 0) red[wave][nt * 16 + lrow] = partial;
  }
  __syncthreads();
  if (t < 64) {
    float s = 0.f;
    #pragma unroll
    for (int w = 0; w < 8; ++w) s += red[w][t];
    out[(size_t)j * DIM + i0 + t] = s;
  }
}

extern "C" void kernel_launch(void* const* d_in, const int* in_sizes, int n_in,
                              void* d_out, int out_size, void* d_ws, size_t ws_size,
                              hipStream_t stream) {
  (void)in_sizes; (void)n_in; (void)out_size; (void)ws_size;
  const float* in_real = (const float*)d_in[0];
  const float* in_imag = (const float*)d_in[1];
  const float* kern    = (const float*)d_in[2];
  float* out = (float*)d_out;

  __bf16* ws  = (__bf16*)d_ws;           // needs 4 * 524288 * 2B = 4 MB scratch
  __bf16* bph = ws;                      // [mat][i][k], mats contiguous in this order
  __bf16* bpl = ws + (size_t)1 * NB;
  __bf16* bqh = ws + (size_t)2 * NB;
  __bf16* bql = ws + (size_t)3 * NB;

  prep_kernel<<<NB / 256, 256, 0, stream>>>(kern, bph, bpl, bqh, bql);
  // i-tiles fastest so the 8 blocks sharing one batch's 2MB A-panel co-run (L2 reuse)
  quad_kernel<<<dim3(8, 128), 256, 0, stream>>>(in_real, in_imag, kern, ws, out);
}
